// FullyConnectedTensorProduct_59236188946488
// MI455X (gfx1250) — compile-verified, matrix operand order checked by reading, not run on hardware
//
#include <hip/hip_runtime.h>

// ---------------------------------------------------------------------------
// FullyConnectedTensorProduct for MI455X (gfx1250, wave32, WMMA bf16)
//
// out_s[z,w]   = A0*( sum_{u,v} s1[z,u] s2[z,v] Wss[u,v,w]
//                   + 1/sqrt3 * sum_{u,v,i} v1[z,u,i] v2[z,v,i] Wvv[u,v,w] )
// out_v[z,w,k] = (1/256)*( sum_{u,v} s1[z,u] v2[z,v,k] Wsv[u,v,w]
//                        + sum_{u,v} v1[z,u,k] s2[z,v] Wvs[u,v,w] )
//
// Weights pre-transposed to Wt[w][k] bf16 in d_ws -> WMMA A operand with
// contiguous 16B loads.  B operand = pure bf16 feature fragments cached in
// registers; the per-(z,u) rank-1 scalar is applied as an f32 FMA on the
// 8-chunk WMMA partial sum (valid because each lane's D column is one z).
// Each wave processes 2 z-tiles so every A fragment feeds 2 WMMAs.
// ---------------------------------------------------------------------------

typedef __attribute__((ext_vector_type(16))) __bf16 v16bf;
typedef __attribute__((ext_vector_type(8)))  __bf16 bf16x8;
typedef __attribute__((ext_vector_type(8)))  float  v8f;
typedef __attribute__((ext_vector_type(4)))  float  f32x4;

#define MUL_S 256
#define MUL_V 128
#define DIM   640
#define NB    4096

#define INV_SQRT3 0.57735026918962576f
#define A0_SCALE  0.0034938562f          /* sqrt(1/81920) */
#define V_SCALE   0.00390625f            /* A1/sqrt3 = 1/256 */

// ws layout (bf16 elements)
#define WS_SS_OFF 0u
#define WS_VV_OFF 16777216u
#define WS_SV_OFF 20971520u
#define WS_VS_OFF 25165824u

// padded LDS row strides (elements) -> conflict-free row-major lane reads
#define S1_STR 257   /* f32  */
#define V1_STR 129   /* f32  */
#define S2_STR 264   /* bf16 */
#define V2_STR 136   /* bf16 */

// ---------------------------------------------------------------------------
// Prep: tiled transpose + f32->bf16 convert.  in: [K][NW] f32, out: [NW][K] bf16
// ---------------------------------------------------------------------------
__global__ void __launch_bounds__(1024)
transpose_cvt_kernel(const float* __restrict__ in, __bf16* __restrict__ out,
                     int K, int NW) {
  __shared__ float tile[32][33];
  int k0 = blockIdx.x * 32;
  int w0 = blockIdx.y * 32;
  int tx = threadIdx.x, ty = threadIdx.y;
  tile[ty][tx] = in[(size_t)(k0 + ty) * NW + (w0 + tx)];
  __syncthreads();
  out[(size_t)(w0 + ty) * K + (k0 + tx)] = (__bf16)tile[tx][ty];
}

// ---------------------------------------------------------------------------
// Fragment helpers.  16-bit operand layout (CDNA5 ISA 7.12.2), lane L:
//   mn = L&15 (A row / B col), half = L>>4
//   element e<8  -> k = 8*half + e ;  e>=8 -> k = 16 + 8*half + (e-8)
// ---------------------------------------------------------------------------
__device__ __forceinline__ v16bf make_frag16(const __bf16* __restrict__ p, int half) {
  bf16x8 r0 = *(const bf16x8*)(p + 8 * half);
  bf16x8 r1 = *(const bf16x8*)(p + 16 + 8 * half);
  v16bf a;
#pragma unroll
  for (int e = 0; e < 8; ++e) { a[e] = r0[e]; a[8 + e] = r1[e]; }
  return a;
}

__device__ __forceinline__ v8f wmma_bf16(v16bf a, v16bf b, v8f c) {
  return __builtin_amdgcn_wmma_f32_16x16x32_bf16(false, a, false, b, (short)0, c,
                                                 false, false);
}

__device__ __forceinline__ void fma_acc(v8f& acc, float s, v8f t) {
#pragma unroll
  for (int r = 0; r < 8; ++r) acc[r] += s * t[r];
}

#define ZERO8 {0.f, 0.f, 0.f, 0.f, 0.f, 0.f, 0.f, 0.f}

// ---------------------------------------------------------------------------
// Kernel S: scalar outputs.  grid (B/32, 2), block 256 (8 waves).
// Block: z-tiles [z0, z0+32).  Wave: w-tile of 16, both z-tiles.
// ---------------------------------------------------------------------------
__global__ void __launch_bounds__(256)
tp_out_s_kernel(const float* __restrict__ x1, const float* __restrict__ x2,
                const __bf16* __restrict__ wt_ss, const __bf16* __restrict__ wt_vv,
                float* __restrict__ out) {
  extern __shared__ char smem[];
  float*  s1f = (float*)smem;                 // [32][257] f32
  float*  v1f = (float*)(smem + 32896);       // [3][32][129] f32 (pre * 1/sqrt3)
  __bf16* s2b = (__bf16*)(smem + 82432);      // [32][264] bf16
  __bf16* v2b = (__bf16*)(smem + 99328);      // [3][32][136] bf16   (end 125440)

  const int z0  = blockIdx.x * 32;
  const int tid = threadIdx.x;

  for (int t = tid; t < 8192; t += 256) {
    int r = t >> 8, c = t & 255;
    s1f[r * S1_STR + c] = x1[(size_t)(z0 + r) * DIM + c];
    s2b[r * S2_STR + c] = (__bf16)x2[(size_t)(z0 + r) * DIM + c];
  }
  for (int t = tid; t < 12288; t += 256) {
    int i = t >> 12, rem = t & 4095, r = rem >> 7, m = rem & 127;
    size_t src = (size_t)(z0 + r) * DIM + MUL_S + m * 3 + i;
    v1f[i * 32 * V1_STR + r * V1_STR + m] = INV_SQRT3 * x1[src];
    v2b[i * 32 * V2_STR + r * V2_STR + m] = (__bf16)x2[src];
  }
  __syncthreads();

  const int wave = tid >> 5, lane = tid & 31;
  const int half = lane >> 4, mn = lane & 15;
  const int w0 = blockIdx.y * 128 + wave * 16;

  const __bf16* wrow_ss = wt_ss + (size_t)(w0 + mn) * 65536;
  const __bf16* wrow_vv = wt_vv + (size_t)(w0 + mn) * 16384;

  v8f acc0 = ZERO8, acc1 = ZERO8;

  // ---- SS: cache 2x8 pure-s2 B fragments in registers --------------------
  {
    v16bf bs[2][8];
#pragma unroll
    for (int zt = 0; zt < 2; ++zt)
#pragma unroll
      for (int c = 0; c < 8; ++c)
        bs[zt][c] = make_frag16(s2b + (zt * 16 + mn) * S2_STR + c * 32, half);

    for (int u = 0; u < 256; ++u) {
      const __bf16* wk = wrow_ss + (size_t)u * 256;
      __builtin_prefetch(wk + 512, 0, 0);
      const float sv0 = s1f[mn * S1_STR + u];
      const float sv1 = s1f[(16 + mn) * S1_STR + u];
      v8f t0 = ZERO8, t1 = ZERO8;
#pragma unroll
      for (int c = 0; c < 8; ++c) {
        v16bf A = make_frag16(wk + c * 32, half);
        t0 = wmma_bf16(A, bs[0][c], t0);
        t1 = wmma_bf16(A, bs[1][c], t1);
      }
      fma_acc(acc0, sv0, t0);
      fma_acc(acc1, sv1, t1);
    }
  }

  // ---- VV: component-major so B fragments stay register-resident ---------
  for (int i = 0; i < 3; ++i) {
    v16bf bv[2][4];
#pragma unroll
    for (int zt = 0; zt < 2; ++zt)
#pragma unroll
      for (int c = 0; c < 4; ++c)
        bv[zt][c] =
            make_frag16(v2b + i * 32 * V2_STR + (zt * 16 + mn) * V2_STR + c * 32, half);

    for (int u = 0; u < 128; ++u) {
      const __bf16* wk = wrow_vv + (size_t)u * 128;
      __builtin_prefetch(wk + 256, 0, 0);
      const float a0 = v1f[i * 32 * V1_STR + mn * V1_STR + u];
      const float a1 = v1f[i * 32 * V1_STR + (16 + mn) * V1_STR + u];
      v8f t0 = ZERO8, t1 = ZERO8;
#pragma unroll
      for (int c = 0; c < 4; ++c) {
        v16bf A = make_frag16(wk + c * 32, half);
        t0 = wmma_bf16(A, bv[0][c], t0);
        t1 = wmma_bf16(A, bv[1][c], t1);
      }
      fma_acc(acc0, a0, t0);
      fma_acc(acc1, a1, t1);
    }
  }

  // D: lane (mn, half), VGPR r -> out[z][w0 + 8*half + r]
  float* o0 = out + (size_t)(z0 + mn) * DIM + w0 + 8 * half;
  float* o1 = out + (size_t)(z0 + 16 + mn) * DIM + w0 + 8 * half;
  f32x4 a = {A0_SCALE * acc0[0], A0_SCALE * acc0[1], A0_SCALE * acc0[2], A0_SCALE * acc0[3]};
  f32x4 b = {A0_SCALE * acc0[4], A0_SCALE * acc0[5], A0_SCALE * acc0[6], A0_SCALE * acc0[7]};
  f32x4 c = {A0_SCALE * acc1[0], A0_SCALE * acc1[1], A0_SCALE * acc1[2], A0_SCALE * acc1[3]};
  f32x4 d = {A0_SCALE * acc1[4], A0_SCALE * acc1[5], A0_SCALE * acc1[6], A0_SCALE * acc1[7]};
  *(f32x4*)(o0)     = a;
  *(f32x4*)(o0 + 4) = b;
  *(f32x4*)(o1)     = c;
  *(f32x4*)(o1 + 4) = d;
}

// ---------------------------------------------------------------------------
// Kernel V: vector outputs.  grid (B/32, 1, 3), block 256 (8 waves -> w 0..127)
// ---------------------------------------------------------------------------
__global__ void __launch_bounds__(256)
tp_out_v_kernel(const float* __restrict__ x1, const float* __restrict__ x2,
                const __bf16* __restrict__ wt_sv, const __bf16* __restrict__ wt_vs,
                float* __restrict__ out) {
  extern __shared__ char smem[];
  float*  s1f  = (float*)smem;                // [32][257] f32
  float*  v1kf = (float*)(smem + 32896);      // [32][129] f32 (component kd)
  __bf16* s2b  = (__bf16*)(smem + 49408);     // [32][264] bf16
  __bf16* v2kb = (__bf16*)(smem + 66304);     // [32][136] bf16   (end 75008)

  const int z0  = blockIdx.x * 32;
  const int kd  = blockIdx.z;
  const int tid = threadIdx.x;

  for (int t = tid; t < 8192; t += 256) {
    int r = t >> 8, c = t & 255;
    s1f[r * S1_STR + c] = x1[(size_t)(z0 + r) * DIM + c];
    s2b[r * S2_STR + c] = (__bf16)x2[(size_t)(z0 + r) * DIM + c];
  }
  for (int t = tid; t < 4096; t += 256) {
    int r = t >> 7, m = t & 127;
    size_t src = (size_t)(z0 + r) * DIM + MUL_S + m * 3 + kd;
    v1kf[r * V1_STR + m] = x1[src];
    v2kb[r * V2_STR + m] = (__bf16)x2[src];
  }
  __syncthreads();

  const int wave = tid >> 5, lane = tid & 31;
  const int half = lane >> 4, mn = lane & 15;
  const int w0 = wave * 16;

  const __bf16* wrow_sv = wt_sv + (size_t)(w0 + mn) * 32768;
  const __bf16* wrow_vs = wt_vs + (size_t)(w0 + mn) * 32768;

  v8f acc0 = ZERO8, acc1 = ZERO8;

  // ---- SV: B = v2k fragments (2x4 cached), scalar = s1[z,u] --------------
  {
    v16bf bv[2][4];
#pragma unroll
    for (int zt = 0; zt < 2; ++zt)
#pragma unroll
      for (int c = 0; c < 4; ++c)
        bv[zt][c] = make_frag16(v2kb + (zt * 16 + mn) * V2_STR + c * 32, half);

    for (int u = 0; u < 256; ++u) {
      const __bf16* wk = wrow_sv + (size_t)u * 128;
      __builtin_prefetch(wk + 256, 0, 0);
      const float sv0 = s1f[mn * S1_STR + u];
      const float sv1 = s1f[(16 + mn) * S1_STR + u];
      v8f t0 = ZERO8, t1 = ZERO8;
#pragma unroll
      for (int c = 0; c < 4; ++c) {
        v16bf A = make_frag16(wk + c * 32, half);
        t0 = wmma_bf16(A, bv[0][c], t0);
        t1 = wmma_bf16(A, bv[1][c], t1);
      }
      fma_acc(acc0, sv0, t0);
      fma_acc(acc1, sv1, t1);
    }
  }

  // ---- VS: B = s2 fragments (2x8 cached), scalar = v1k[z,u] --------------
  {
    v16bf bs[2][8];
#pragma unroll
    for (int zt = 0; zt < 2; ++zt)
#pragma unroll
      for (int c = 0; c < 8; ++c)
        bs[zt][c] = make_frag16(s2b + (zt * 16 + mn) * S2_STR + c * 32, half);

    for (int u = 0; u < 128; ++u) {
      const __bf16* wk = wrow_vs + (size_t)u * 256;
      __builtin_prefetch(wk + 512, 0, 0);
      const float vv0 = v1kf[mn * V1_STR + u];
      const float vv1 = v1kf[(16 + mn) * V1_STR + u];
      v8f t0 = ZERO8, t1 = ZERO8;
#pragma unroll
      for (int c = 0; c < 8; ++c) {
        v16bf A = make_frag16(wk + c * 32, half);
        t0 = wmma_bf16(A, bs[0][c], t0);
        t1 = wmma_bf16(A, bs[1][c], t1);
      }
      fma_acc(acc0, vv0, t0);
      fma_acc(acc1, vv1, t1);
    }
  }

  // out[z, 256 + (w*3 + kd)], w = w0 + 8*half + r (stride-3 scalar stores)
  float* ob0 = out + (size_t)(z0 + mn) * DIM + MUL_S + kd;
  float* ob1 = out + (size_t)(z0 + 16 + mn) * DIM + MUL_S + kd;
#pragma unroll
  for (int r = 0; r < 8; ++r) {
    ob0[(size_t)(w0 + 8 * half + r) * 3] = V_SCALE * acc0[r];
    ob1[(size_t)(w0 + 8 * half + r) * 3] = V_SCALE * acc1[r];
  }
}

// ---------------------------------------------------------------------------
extern "C" void kernel_launch(void* const* d_in, const int* in_sizes, int n_in,
                              void* d_out, int out_size, void* d_ws, size_t ws_size,
                              hipStream_t stream) {
  const float* x1    = (const float*)d_in[0];
  const float* x2    = (const float*)d_in[1];
  const float* W_ss0 = (const float*)d_in[2];  // [256][256][256]
  const float* W_vv0 = (const float*)d_in[3];  // [128][128][256]
  const float* W_sv1 = (const float*)d_in[4];  // [256][128][128]
  const float* W_vs1 = (const float*)d_in[5];  // [128][256][128]
  float* out = (float*)d_out;

  __bf16* ws    = (__bf16*)d_ws;               // needs 58,720,256 bytes
  __bf16* wt_ss = ws + WS_SS_OFF;              // [256][65536]
  __bf16* wt_vv = ws + WS_VV_OFF;              // [256][16384]
  __bf16* wt_sv = ws + WS_SV_OFF;              // [128][32768]
  __bf16* wt_vs = ws + WS_VS_OFF;              // [128][32768]

  dim3 tb(32, 32);
  transpose_cvt_kernel<<<dim3(65536 / 32, 256 / 32), tb, 0, stream>>>(W_ss0, wt_ss, 65536, 256);
  transpose_cvt_kernel<<<dim3(16384 / 32, 256 / 32), tb, 0, stream>>>(W_vv0, wt_vv, 16384, 256);
  transpose_cvt_kernel<<<dim3(32768 / 32, 128 / 32), tb, 0, stream>>>(W_sv1, wt_sv, 32768, 128);
  transpose_cvt_kernel<<<dim3(32768 / 32, 128 / 32), tb, 0, stream>>>(W_vs1, wt_vs, 32768, 128);

  tp_out_s_kernel<<<dim3(NB / 32, 2, 1), 256, 125440, stream>>>(x1, x2, wt_ss, wt_vv, out);
  tp_out_v_kernel<<<dim3(NB / 32, 1, 3), 256, 75008, stream>>>(x1, x2, wt_sv, wt_vs, out);
}
